// MultiRelationalHybridGATv2_35296041238630
// MI455X (gfx1250) — matrix operand order repaired and checked by reading, not compiled, not run on hardware
//
#include <hip/hip_runtime.h>
#include <hip/hip_bf16.h>
#include <math.h>

// =====================================================================
// MI455X (gfx1250) implementation of MultiRelationalHybridGATv2.
//
// Precision: all dense GEMMs use bf16 WMMA (v_wmma_f32_16x16x32_bf16)
// with fp32 accumulation; edge projections stored bf16; softmax and
// scatter-adds in fp32.
//
// Workspace layout (~430 MB):
//   3 node buffers N*128 f32, h1/h2/h3 N*128 f32, eproj (E+N)*128 bf16,
//   logits (E+N)*8 f32, nodemax/nodesum N*8 f32, deg N f32.
// =====================================================================

typedef __bf16 bf16_t;
typedef __attribute__((ext_vector_type(16))) __bf16 v16bf;
typedef __attribute__((ext_vector_type(8)))  __bf16 v8bf;
typedef __attribute__((ext_vector_type(4)))  __bf16 v4bf;
typedef __attribute__((ext_vector_type(8)))  float  v8f;

#define KP128 136   // padded K stride (elems) for K=128 tiles -> 272B rows, conflict free
#define KP64  72
#define KP32  40

static __device__ __forceinline__ bf16_t f2bf(float f) {
  unsigned u = __float_as_uint(f);
  unsigned r = (u + 0x7FFFu + ((u >> 16) & 1u)) >> 16;   // round to nearest even
  unsigned short s = (unsigned short)r;
  return __builtin_bit_cast(bf16_t, s);
}
static __device__ __forceinline__ float bf2f(bf16_t b) {
  unsigned short s = __builtin_bit_cast(unsigned short, b);
  return __uint_as_float(((unsigned)s) << 16);
}

static __device__ __forceinline__ void atomicMaxFloat(float* addr, float val) {
  if (val >= 0.f) atomicMax((int*)addr, __float_as_int(val));
  else            atomicMin((unsigned int*)addr, __float_as_uint(val));
}

// ---------------------------------------------------------------------
// WMMA inner product over K (= NS*32) for one 16x16 output tile.
// Software-pipelined: load ALL fragments first (batched ds_load_b128,
// single dscnt wait), then issue the NS WMMAs back-to-back.
// arow: per-lane pointer to LDS row (m0+ln)*stride of A  [row][k] bf16
// brow: per-lane pointer to LDS row (n0+ln)*stride of B^T [n][k]  bf16
// A frag (16x32 bf16): lane(ln,h): elems 0..7 = K 8h..8h+7, 8..15 = K 16+8h..+7
// B frag (32x16 bf16): lane(ln,h): elems 0..7 = K 8h..8h+7, 8..15 = K 16+8h..+7
// ---------------------------------------------------------------------
template <int NS>
static __device__ __forceinline__ v8f wmma_acc_kt(const bf16_t* __restrict__ arow,
                                                  const bf16_t* __restrict__ brow,
                                                  v8f acc) {
  const int lane = threadIdx.x & 31;
  const int h    = lane >> 4;
  v16bf a[NS], b[NS];
#pragma unroll
  for (int s = 0; s < NS; ++s) {
    const int k0 = 32 * s;
    v8bf alo = *(const v8bf*)(arow + k0 + 8 * h);
    v8bf ahi = *(const v8bf*)(arow + k0 + 16 + 8 * h);
    v8bf blo = *(const v8bf*)(brow + k0 + 8 * h);
    v8bf bhi = *(const v8bf*)(brow + k0 + 16 + 8 * h);
#pragma unroll
    for (int i = 0; i < 8; ++i) {
      a[s][i] = alo[i]; a[s][i + 8] = ahi[i];
      b[s][i] = blo[i]; b[s][i + 8] = bhi[i];
    }
  }
#pragma unroll
  for (int s = 0; s < NS; ++s)
    acc = __builtin_amdgcn_wmma_f32_16x16x32_bf16(false, a[s], false, b[s],
                                                  (short)0, acc, false, false);
  return acc;
}

// ---------------------------------------------------------------------
// out[M,128] = A[M,128] @ W[128,128] + bias   (optional relu)
// 64-row tile per block, 256 threads (8 waves), 32 WMMA tiles per block.
// Interior blocks take a branch-free fast path for loads and stores.
// ---------------------------------------------------------------------
__global__ void __launch_bounds__(256)
node_gemm_kernel(const float* __restrict__ A, const float* __restrict__ W,
                 const float* __restrict__ bias, float* __restrict__ out,
                 int M, int relu) {
  __shared__ __align__(16) bf16_t As[64 * KP128];   // 17408 B
  __shared__ __align__(16) bf16_t Bt[128 * KP128];  // 34816 B
  const int m_base = blockIdx.x * 64;
  const int t = threadIdx.x;
  const bool interior = (m_base + 64 <= M);
  // A tile: 8 rows per iter (256 thr * float4 = 1024 elems)
  if (interior) {
#pragma unroll
    for (int it = 0; it < 8; ++it) {
      int r = it * 8 + (t >> 5), c = (t & 31) * 4;
      float4 v = ((const float4*)A)[(size_t)(m_base + r) * 32 + (t & 31)];
      bf16_t* p = &As[r * KP128 + c];
      p[0] = f2bf(v.x); p[1] = f2bf(v.y); p[2] = f2bf(v.z); p[3] = f2bf(v.w);
    }
  } else {
#pragma unroll
    for (int it = 0; it < 8; ++it) {
      int r = it * 8 + (t >> 5), c = (t & 31) * 4;
      int row = m_base + r;
      float4 v = (row < M) ? ((const float4*)A)[(size_t)row * 32 + (t & 31)]
                           : make_float4(0.f, 0.f, 0.f, 0.f);
      bf16_t* p = &As[r * KP128 + c];
      p[0] = f2bf(v.x); p[1] = f2bf(v.y); p[2] = f2bf(v.z); p[3] = f2bf(v.w);
    }
  }
  // W (row-major [k][n]) -> Bt[n][k]
  for (int i = t; i < 128 * 128; i += 256) {
    int k = i >> 7, n = i & 127;
    Bt[n * KP128 + k] = f2bf(W[i]);
  }
  __syncthreads();
  const int wave = t >> 5, lane = t & 31, ln = lane & 15, h = lane >> 4;
  // tile = wave + 8*i -> mt = wave&3 fixed per wave => A fragments reused.
  for (int tile = wave; tile < 32; tile += 8) {
    int mt = tile & 3, nt = tile >> 2;
    v8f acc = {0.f, 0.f, 0.f, 0.f, 0.f, 0.f, 0.f, 0.f};
    acc = wmma_acc_kt<4>(&As[(mt * 16 + ln) * KP128], &Bt[(nt * 16 + ln) * KP128], acc);
    int col = nt * 16 + ln;
    float bb = bias ? bias[col] : 0.f;
    if (interior) {
#pragma unroll
      for (int r = 0; r < 8; ++r) {
        int row = m_base + mt * 16 + r + 8 * h;   // C layout: M = r + 8h, N = ln
        float vv = acc[r] + bb;
        if (relu) vv = fmaxf(vv, 0.f);
        out[(size_t)row * 128 + col] = vv;
      }
    } else {
#pragma unroll
      for (int r = 0; r < 8; ++r) {
        int row = m_base + mt * 16 + r + 8 * h;
        if (row < M) {
          float vv = acc[r] + bb;
          if (relu) vv = fmaxf(vv, 0.f);
          out[(size_t)row * 128 + col] = vv;
        }
      }
    }
  }
}

// ---------------------------------------------------------------------
// Fused edge MLP + projection:
//   eproj[e] = relu(relu(ea[e]@W1+b1)@W2+b2)@We (+be)   -> bf16 [E,128]
// 64-edge tile; three WMMA stages fully LDS-resident (60KB).
// ---------------------------------------------------------------------
__global__ void __launch_bounds__(256)
edge_mlp_kernel(const float* __restrict__ ea,
                const float* __restrict__ W1, const float* __restrict__ b1,
                const float* __restrict__ W2, const float* __restrict__ b2,
                const float* __restrict__ We, const float* __restrict__ be,
                bf16_t* __restrict__ eproj, int E) {
  __shared__ __align__(16) bf16_t h1s[64 * KP64];     // 9216 B
  __shared__ __align__(16) bf16_t h2s[64 * KP128];    // 17408 B
  __shared__ __align__(16) bf16_t scr[128 * KP128];   // 34816 B (reused per stage)
  bf16_t* eas = scr;                 // 64 x KP32
  bf16_t* W1t = scr + 64 * KP32;     // 64 x KP32
  bf16_t* W2t = scr;                 // 128 x KP64
  bf16_t* Wet = scr;                 // 128 x KP128
  const int e_base = blockIdx.x * 64;
  const int t = threadIdx.x, wave = t >> 5, lane = t & 31, ln = lane & 15, h = lane >> 4;

  // ---- stage 1: ea(64x16, K padded to 32) @ W1(16x64) -> h1s
  for (int i = t; i < 2 * 64 * KP32; i += 256) scr[i] = f2bf(0.f);  // zero eas+W1t
  __syncthreads();
  {
    int r = t >> 2, c4 = t & 3;
    int e = e_base + r;
    if (e < E) {
      float4 v = ((const float4*)ea)[(size_t)e * 4 + c4];
      bf16_t* p = &eas[r * KP32 + c4 * 4];
      p[0] = f2bf(v.x); p[1] = f2bf(v.y); p[2] = f2bf(v.z); p[3] = f2bf(v.w);
    }
  }
  for (int i = t; i < 16 * 64; i += 256) {
    int k = i >> 6, n = i & 63;
    W1t[n * KP32 + k] = f2bf(W1[i]);
  }
  __syncthreads();
  for (int tile = wave; tile < 16; tile += 8) {
    int mt = tile & 3, nt = tile >> 2;
    v8f acc = {0.f, 0.f, 0.f, 0.f, 0.f, 0.f, 0.f, 0.f};
    acc = wmma_acc_kt<1>(&eas[(mt * 16 + ln) * KP32], &W1t[(nt * 16 + ln) * KP32], acc);
    int col = nt * 16 + ln;
    float bb = b1[col];
#pragma unroll
    for (int r = 0; r < 8; ++r)
      h1s[(mt * 16 + r + 8 * h) * KP64 + col] = f2bf(fmaxf(acc[r] + bb, 0.f));
  }
  __syncthreads();
  // ---- stage 2: h1(64x64) @ W2(64x128) -> h2s
  for (int i = t; i < 64 * 128; i += 256) {
    int k = i >> 7, n = i & 127;
    W2t[n * KP64 + k] = f2bf(W2[i]);
  }
  __syncthreads();
  for (int tile = wave; tile < 32; tile += 8) {
    int mt = tile & 3, nt = tile >> 2;
    v8f acc = {0.f, 0.f, 0.f, 0.f, 0.f, 0.f, 0.f, 0.f};
    acc = wmma_acc_kt<2>(&h1s[(mt * 16 + ln) * KP64], &W2t[(nt * 16 + ln) * KP64], acc);
    int col = nt * 16 + ln;
    float bb = b2[col];
#pragma unroll
    for (int r = 0; r < 8; ++r)
      h2s[(mt * 16 + r + 8 * h) * KP128 + col] = f2bf(fmaxf(acc[r] + bb, 0.f));
  }
  __syncthreads();
  // ---- stage 3: h2(64x128) @ We(128x128) -> eproj (bf16)
  for (int i = t; i < 128 * 128; i += 256) {
    int k = i >> 7, n = i & 127;
    Wet[n * KP128 + k] = f2bf(We[i]);
  }
  __syncthreads();
  for (int tile = wave; tile < 32; tile += 8) {
    int mt = tile & 3, nt = tile >> 2;
    v8f acc = {0.f, 0.f, 0.f, 0.f, 0.f, 0.f, 0.f, 0.f};
    acc = wmma_acc_kt<4>(&h2s[(mt * 16 + ln) * KP128], &Wet[(nt * 16 + ln) * KP128], acc);
    int col = nt * 16 + ln;
    float bb = be ? be[col] : 0.f;
#pragma unroll
    for (int r = 0; r < 8; ++r) {
      int e = e_base + mt * 16 + r + 8 * h;
      if (e < E) eproj[(size_t)e * 128 + col] = f2bf(acc[r] + bb);
    }
  }
}

// ---------------------------------------------------------------------
// Fusion: out = relu(concat(h1,h2,h3) @ Wf(384x128) + bf)
// K split into 3 chunks of 128, accumulators held across chunks.
// ---------------------------------------------------------------------
__global__ void __launch_bounds__(256)
fusion_kernel(const float* __restrict__ h1, const float* __restrict__ h2,
              const float* __restrict__ h3, const float* __restrict__ Wf,
              const float* __restrict__ bfu, float* __restrict__ out, int M) {
  __shared__ __align__(16) bf16_t As[64 * KP128];
  __shared__ __align__(16) bf16_t Bt[128 * KP128];
  const float* hs[3] = {h1, h2, h3};
  const int m_base = blockIdx.x * 64;
  const int t = threadIdx.x, wave = t >> 5, lane = t & 31, ln = lane & 15, h = lane >> 4;
  const bool interior = (m_base + 64 <= M);
  v8f acc[4];
#pragma unroll
  for (int ti = 0; ti < 4; ++ti) acc[ti] = (v8f){0.f, 0.f, 0.f, 0.f, 0.f, 0.f, 0.f, 0.f};
  for (int ck = 0; ck < 3; ++ck) {
    const float* A = hs[ck];
    if (interior) {
#pragma unroll
      for (int it = 0; it < 8; ++it) {
        int r = it * 8 + (t >> 5), c = (t & 31) * 4;
        float4 v = ((const float4*)A)[(size_t)(m_base + r) * 32 + (t & 31)];
        bf16_t* p = &As[r * KP128 + c];
        p[0] = f2bf(v.x); p[1] = f2bf(v.y); p[2] = f2bf(v.z); p[3] = f2bf(v.w);
      }
    } else {
#pragma unroll
      for (int it = 0; it < 8; ++it) {
        int r = it * 8 + (t >> 5), c = (t & 31) * 4;
        int row = m_base + r;
        float4 v = (row < M) ? ((const float4*)A)[(size_t)row * 32 + (t & 31)]
                             : make_float4(0.f, 0.f, 0.f, 0.f);
        bf16_t* p = &As[r * KP128 + c];
        p[0] = f2bf(v.x); p[1] = f2bf(v.y); p[2] = f2bf(v.z); p[3] = f2bf(v.w);
      }
    }
    for (int i = t; i < 128 * 128; i += 256) {
      int k = i >> 7, n = i & 127;
      Bt[n * KP128 + k] = f2bf(Wf[(size_t)(ck * 128 + k) * 128 + n]);
    }
    __syncthreads();
#pragma unroll
    for (int ti = 0; ti < 4; ++ti) {
      int tile = wave + ti * 8, mt = tile & 3, nt = tile >> 2;
      acc[ti] = wmma_acc_kt<4>(&As[(mt * 16 + ln) * KP128],
                               &Bt[(nt * 16 + ln) * KP128], acc[ti]);
    }
    __syncthreads();
  }
#pragma unroll
  for (int ti = 0; ti < 4; ++ti) {
    int tile = wave + ti * 8, mt = tile & 3, nt = tile >> 2;
    int col = nt * 16 + ln;
    float bb = bfu[col];
    if (interior) {
#pragma unroll
      for (int r = 0; r < 8; ++r) {
        int row = m_base + mt * 16 + r + 8 * h;
        out[(size_t)row * 128 + col] = fmaxf(acc[ti][r] + bb, 0.f);
      }
    } else {
#pragma unroll
      for (int r = 0; r < 8; ++r) {
        int row = m_base + mt * 16 + r + 8 * h;
        if (row < M) out[(size_t)row * 128 + col] = fmaxf(acc[ti][r] + bb, 0.f);
      }
    }
  }
}

// ---------------------------------------------------------------------
// Per-edge attention logits (one wave per edge; lane l owns channels 4l..4l+3,
// head = l/4). mode 0: GATv2  logit = att . leaky(xl[s]+xr[d]+ep)
//              mode 1: Transformer logit = q[d].(k[s]+ep) / 4
// Edges e >= E are self loops (s=d=e-E), used for r1/r2.
// ---------------------------------------------------------------------
__global__ void __launch_bounds__(256)
edge_logits_kernel(int mode, const float* __restrict__ srcfeat,
                   const float* __restrict__ dstfeat,
                   const bf16_t* __restrict__ eproj, const float* __restrict__ att,
                   const int* __restrict__ src, const int* __restrict__ dst,
                   int E, int nloop, float* __restrict__ logits,
                   float* __restrict__ nodemax) {
  const int widx = blockIdx.x * 8 + (threadIdx.x >> 5);
  const int lane = threadIdx.x & 31;
  const int etot = E + nloop;
  if (widx >= etot) return;
  int s, d;
  if (widx < E) { s = src[widx]; d = dst[widx]; } else { s = d = widx - E; }
  float4 a4 = ((const float4*)(srcfeat + (size_t)s * 128))[lane];
  float4 b4 = ((const float4*)(dstfeat + (size_t)d * 128))[lane];
  float mx = a4.x, my = a4.y, mz = a4.z, mw = a4.w;
  if (eproj) {
    v4bf e4 = ((const v4bf*)(eproj + (size_t)widx * 128))[lane];
    mx += bf2f(e4[0]); my += bf2f(e4[1]); mz += bf2f(e4[2]); mw += bf2f(e4[3]);
  }
  float part;
  if (mode == 0) {
    mx += b4.x; my += b4.y; mz += b4.z; mw += b4.w;
    mx = mx > 0.f ? mx : 0.2f * mx; my = my > 0.f ? my : 0.2f * my;
    mz = mz > 0.f ? mz : 0.2f * mz; mw = mw > 0.f ? mw : 0.2f * mw;
    float4 at = ((const float4*)att)[lane];   // att[h][c%16] == flat[4*lane..]
    part = mx * at.x + my * at.y + mz * at.z + mw * at.w;
  } else {
    part = (mx * b4.x + my * b4.y + mz * b4.z + mw * b4.w) * 0.25f;  // 1/sqrt(16)
  }
  part += __shfl_xor(part, 1, 32);
  part += __shfl_xor(part, 2, 32);
  if ((lane & 3) == 0) {
    int hh = lane >> 2;
    logits[(size_t)widx * 8 + hh] = part;
    atomicMaxFloat(&nodemax[(size_t)d * 8 + hh], part);
  }
}

// exp(logit - max[dst]) in-place + segment sum (thread per (edge,head))
__global__ void __launch_bounds__(256)
seg_exp_kernel(float* __restrict__ logits, const int* __restrict__ dst,
               int E, int nloop, const float* __restrict__ nodemax,
               float* __restrict__ nodesum) {
  const int etot = E + nloop;
  int i = blockIdx.x * 256 + threadIdx.x;
  if (i >= etot * 8) return;
  int e = i >> 3, hh = i & 7;
  int d = (e < E) ? dst[e] : (e - E);
  float ex = expf(logits[i] - nodemax[(size_t)d * 8 + hh]);
  logits[i] = ex;
  atomicAdd(&nodesum[(size_t)d * 8 + hh], ex);
}

// out[dst] += (vfeat[src] (+ eproj)) * alpha    (wave per edge)
__global__ void __launch_bounds__(256)
edge_value_kernel(const float* __restrict__ vfeat, const bf16_t* __restrict__ eproj,
                  const float* __restrict__ ex, const float* __restrict__ nodesum,
                  const int* __restrict__ src, const int* __restrict__ dst,
                  int E, int nloop, float* __restrict__ out) {
  const int widx = blockIdx.x * 8 + (threadIdx.x >> 5);
  const int lane = threadIdx.x & 31;
  const int etot = E + nloop;
  if (widx >= etot) return;
  int s, d;
  if (widx < E) { s = src[widx]; d = dst[widx]; } else { s = d = widx - E; }
  int hh = lane >> 2;
  float alpha = ex[(size_t)widx * 8 + hh] / (nodesum[(size_t)d * 8 + hh] + 1e-16f);
  float4 v4 = ((const float4*)(vfeat + (size_t)s * 128))[lane];
  float vx = v4.x, vy = v4.y, vz = v4.z, vw = v4.w;
  if (eproj) {
    v4bf e4 = ((const v4bf*)(eproj + (size_t)widx * 128))[lane];
    vx += bf2f(e4[0]); vy += bf2f(e4[1]); vz += bf2f(e4[2]); vw += bf2f(e4[3]);
  }
  float* o = out + (size_t)d * 128 + lane * 4;
  atomicAdd(o + 0, vx * alpha);
  atomicAdd(o + 1, vy * alpha);
  atomicAdd(o + 2, vz * alpha);
  atomicAdd(o + 3, vw * alpha);
}

// ---------------------- small utility kernels ------------------------
__global__ void __launch_bounds__(256)
fill_f32_kernel(float* __restrict__ p, int n, float v) {
  int i = blockIdx.x * 256 + threadIdx.x;
  if (i < n) p[i] = v;
}
__global__ void __launch_bounds__(256)
init_bias_kernel(float* __restrict__ out, const float* __restrict__ bias, int n) {
  int i = blockIdx.x * 256 + threadIdx.x;
  if (i < n) out[i] = bias[i & 127];
}
__global__ void __launch_bounds__(256)
deg_kernel(const int* __restrict__ dst, float* __restrict__ deg, int E) {
  int i = blockIdx.x * 256 + threadIdx.x;
  if (i < E) atomicAdd(&deg[dst[i]], 1.f);
}
__global__ void __launch_bounds__(256)
scatter_eproj_kernel(const int* __restrict__ dst, const bf16_t* __restrict__ eproj,
                     float* __restrict__ selfsum, int E) {
  const int widx = blockIdx.x * 8 + (threadIdx.x >> 5);
  const int lane = threadIdx.x & 31;
  if (widx >= E) return;
  int d = dst[widx];
  v4bf e4 = ((const v4bf*)(eproj + (size_t)widx * 128))[lane];
  float* o = selfsum + (size_t)d * 128 + lane * 4;
  atomicAdd(o + 0, bf2f(e4[0]));
  atomicAdd(o + 1, bf2f(e4[1]));
  atomicAdd(o + 2, bf2f(e4[2]));
  atomicAdd(o + 3, bf2f(e4[3]));
}
__global__ void __launch_bounds__(256)
selfloop_mean_kernel(bf16_t* __restrict__ eproj, const float* __restrict__ selfsum,
                     const float* __restrict__ deg, int N, int E) {
  int i = blockIdx.x * 256 + threadIdx.x;
  if (i >= N * 128) return;
  int n = i >> 7;
  eproj[(size_t)(E + n) * 128 + (i & 127)] = f2bf(selfsum[i] / fmaxf(deg[n], 1.f));
}

// =====================================================================
// Host launcher.  Input order = setup_inputs() dict order with params
// flattened in jax pytree (sorted-key) order.
// =====================================================================
extern "C" void kernel_launch(void* const* d_in, const int* in_sizes, int n_in,
                              void* d_out, int out_size, void* d_ws, size_t ws_size,
                              hipStream_t stream) {
  (void)n_in; (void)out_size; (void)ws_size;
  const int N = in_sizes[0] / 128;
  const int E = in_sizes[1] / 2;

  const float* x    = (const float*)d_in[0];
  const int*   ei1  = (const int*)d_in[1];
  const int*   ei2  = (const int*)d_in[2];
  const int*   ei3  = (const int*)d_in[3];
  const float* ea2  = (const float*)d_in[4];
  const float* ea3  = (const float*)d_in[5];
  const float* fW   = (const float*)d_in[6];   // fusion.W (384x128)
  const float* fb   = (const float*)d_in[7];   // fusion.b
  const float* r1Wl = (const float*)d_in[8];
  const float* r1Wr = (const float*)d_in[9];
  const float* r1att  = (const float*)d_in[10];
  const float* r1bias = (const float*)d_in[11];
  const float* r1bl = (const float*)d_in[12];
  const float* r1br = (const float*)d_in[13];
  const float* r2We = (const float*)d_in[14];
  const float* r2Wl = (const float*)d_in[15];
  const float* r2Wr = (const float*)d_in[16];
  const float* r2att  = (const float*)d_in[17];
  const float* r2bias = (const float*)d_in[18];
  const float* r2bl = (const float*)d_in[19];
  const float* r2br = (const float*)d_in[20];
  const float* r2W1 = (const float*)d_in[21];
  const float* r2W2 = (const float*)d_in[22];
  const float* r2b1 = (const float*)d_in[23];
  const float* r2b2 = (const float*)d_in[24];
  const float* r3We = (const float*)d_in[25];
  const float* r3Wk = (const float*)d_in[26];
  const float* r3Wq = (const float*)d_in[27];
  const float* r3Wskip = (const float*)d_in[28];
  const float* r3Wv = (const float*)d_in[29];
  const float* r3be = (const float*)d_in[30];
  const float* r3bk = (const float*)d_in[31];
  const float* r3bq = (const float*)d_in[32];
  const float* r3bskip = (const float*)d_in[33];
  const float* r3bv = (const float*)d_in[34];
  const float* r3W1 = (const float*)d_in[35];
  const float* r3W2 = (const float*)d_in[36];
  const float* r3b1 = (const float*)d_in[37];
  const float* r3b2 = (const float*)d_in[38];

  // ---- workspace carve-up
  char* w = (char*)d_ws;
  auto alloc = [&](size_t bytes) -> char* {
    char* p = w;
    w += (bytes + 255) & ~(size_t)255;
    return p;
  };
  float* nb0 = (float*)alloc((size_t)N * 128 * 4);   // xl / q
  float* nb1 = (float*)alloc((size_t)N * 128 * 4);   // xr / k
  float* nb2 = (float*)alloc((size_t)N * 128 * 4);   // selfsum (r2) / v (r3)
  float* h1  = (float*)alloc((size_t)N * 128 * 4);
  float* h2  = (float*)alloc((size_t)N * 128 * 4);
  float* h3  = (float*)alloc((size_t)N * 128 * 4);
  bf16_t* eproj = (bf16_t*)alloc((size_t)(E + N) * 128 * 2);
  float* logits  = (float*)alloc((size_t)(E + N) * 8 * 4);
  float* nodemax = (float*)alloc((size_t)N * 8 * 4);
  float* nodesum = (float*)alloc((size_t)N * 8 * 4);
  float* deg     = (float*)alloc((size_t)N * 4);

  auto cdiv = [](long a, long b) -> int { return (int)((a + b - 1) / b); };
  const int gB   = cdiv(N, 64);          // GEMM blocks
  const int et12 = E + N;                // edges + self loops

  // ================= relation 1: GATv2, no edge attrs =================
  node_gemm_kernel<<<gB, 256, 0, stream>>>(x, r1Wl, r1bl, nb0, N, 0);
  node_gemm_kernel<<<gB, 256, 0, stream>>>(x, r1Wr, r1br, nb1, N, 0);
  fill_f32_kernel<<<cdiv(N * 8, 256), 256, 0, stream>>>(nodemax, N * 8, -INFINITY);
  fill_f32_kernel<<<cdiv(N * 8, 256), 256, 0, stream>>>(nodesum, N * 8, 0.f);
  init_bias_kernel<<<cdiv((long)N * 128, 256), 256, 0, stream>>>(h1, r1bias, N * 128);
  edge_logits_kernel<<<cdiv(et12, 8), 256, 0, stream>>>(
      0, nb0, nb1, (const bf16_t*)nullptr, r1att, ei1, ei1 + E, E, N, logits, nodemax);
  seg_exp_kernel<<<cdiv((long)et12 * 8, 256), 256, 0, stream>>>(
      logits, ei1 + E, E, N, nodemax, nodesum);
  edge_value_kernel<<<cdiv(et12, 8), 256, 0, stream>>>(
      nb0, (const bf16_t*)nullptr, logits, nodesum, ei1, ei1 + E, E, N, h1);

  // ================= relation 2: edge-MLP GATv2 =======================
  edge_mlp_kernel<<<cdiv(E, 64), 256, 0, stream>>>(ea2, r2W1, r2b1, r2W2, r2b2,
                                                   r2We, (const float*)nullptr,
                                                   eproj, E);
  fill_f32_kernel<<<cdiv(N, 256), 256, 0, stream>>>(deg, N, 0.f);
  fill_f32_kernel<<<cdiv((long)N * 128, 256), 256, 0, stream>>>(nb2, N * 128, 0.f);
  deg_kernel<<<cdiv(E, 256), 256, 0, stream>>>(ei2 + E, deg, E);
  scatter_eproj_kernel<<<cdiv(E, 8), 256, 0, stream>>>(ei2 + E, eproj, nb2, E);
  selfloop_mean_kernel<<<cdiv((long)N * 128, 256), 256, 0, stream>>>(eproj, nb2, deg, N, E);
  node_gemm_kernel<<<gB, 256, 0, stream>>>(x, r2Wl, r2bl, nb0, N, 0);
  node_gemm_kernel<<<gB, 256, 0, stream>>>(x, r2Wr, r2br, nb1, N, 0);
  fill_f32_kernel<<<cdiv(N * 8, 256), 256, 0, stream>>>(nodemax, N * 8, -INFINITY);
  fill_f32_kernel<<<cdiv(N * 8, 256), 256, 0, stream>>>(nodesum, N * 8, 0.f);
  init_bias_kernel<<<cdiv((long)N * 128, 256), 256, 0, stream>>>(h2, r2bias, N * 128);
  edge_logits_kernel<<<cdiv(et12, 8), 256, 0, stream>>>(
      0, nb0, nb1, eproj, r2att, ei2, ei2 + E, E, N, logits, nodemax);
  seg_exp_kernel<<<cdiv((long)et12 * 8, 256), 256, 0, stream>>>(
      logits, ei2 + E, E, N, nodemax, nodesum);
  edge_value_kernel<<<cdiv(et12, 8), 256, 0, stream>>>(
      nb0, (const bf16_t*)nullptr, logits, nodesum, ei2, ei2 + E, E, N, h2);

  // ================= relation 3: TransformerConv ======================
  edge_mlp_kernel<<<cdiv(E, 64), 256, 0, stream>>>(ea3, r3W1, r3b1, r3W2, r3b2,
                                                   r3We, r3be, eproj, E);
  node_gemm_kernel<<<gB, 256, 0, stream>>>(x, r3Wq, r3bq, nb0, N, 0);  // q
  node_gemm_kernel<<<gB, 256, 0, stream>>>(x, r3Wk, r3bk, nb1, N, 0);  // k
  node_gemm_kernel<<<gB, 256, 0, stream>>>(x, r3Wv, r3bv, nb2, N, 0);  // v
  node_gemm_kernel<<<gB, 256, 0, stream>>>(x, r3Wskip, r3bskip, h3, N, 0);  // skip
  fill_f32_kernel<<<cdiv(N * 8, 256), 256, 0, stream>>>(nodemax, N * 8, -INFINITY);
  fill_f32_kernel<<<cdiv(N * 8, 256), 256, 0, stream>>>(nodesum, N * 8, 0.f);
  edge_logits_kernel<<<cdiv(E, 8), 256, 0, stream>>>(
      1, nb1, nb0, eproj, (const float*)nullptr, ei3, ei3 + E, E, 0, logits, nodemax);
  seg_exp_kernel<<<cdiv((long)E * 8, 256), 256, 0, stream>>>(
      logits, ei3 + E, E, 0, nodemax, nodesum);
  edge_value_kernel<<<cdiv(E, 8), 256, 0, stream>>>(
      nb2, eproj, logits, nodesum, ei3, ei3 + E, E, 0, h3);

  // ================= fusion ===========================================
  fusion_kernel<<<gB, 256, 0, stream>>>(h1, h2, h3, fW, fb, (float*)d_out, N);
}
